// ChunkStickyRouter_57226144252185
// MI455X (gfx1250) — matrix operand order repaired and checked by reading, not compiled, not run on hardware
//
#include <hip/hip_runtime.h>
#include <math.h>

typedef __bf16 bf16_t;
typedef __attribute__((ext_vector_type(16))) __bf16 v16bf;
typedef __attribute__((ext_vector_type(8)))  __bf16 v8bf;
typedef __attribute__((ext_vector_type(8)))  float  v8f;
typedef __attribute__((ext_vector_type(4)))  float  v4f;

#define LDT 40           // padded LDS K-stride (elements); 40*2B = 80B, 16B aligned
#define CHUNK_SZ 128
#define TAU_C 0.7f

__device__ __forceinline__ bf16_t f2bf(float f) { return (bf16_t)f; }

// Low 32 bits of a generic pointer to __shared__ = wave LDS byte offset
// (ISA 10.2: LDS aperture).
__device__ __forceinline__ unsigned ldsOff(const void* p) {
    return (unsigned)(uintptr_t)p;
}

// async 16B copy: global -> LDS, tracked by ASYNCcnt (ISA 08 / 15.18.3 op 98)
__device__ __forceinline__ void asyncCopyB128(unsigned lds, uint64_t gaddr) {
    asm volatile("global_load_async_to_lds_b128 %0, %1, off"
                 :: "v"(lds), "v"(gaddr) : "memory");
}

// ---------------------------------------------------------------------------
// elementwise fp32 -> bf16 (8 elems/thread, b128 stores)
// ---------------------------------------------------------------------------
__global__ __launch_bounds__(256)
void cvt_bf16_kernel(const float* __restrict__ in, bf16_t* __restrict__ out, long n)
{
    long i = ((long)blockIdx.x * 256 + threadIdx.x) * 8;
    if (i + 8 > n) return;
    v4f a = *(const v4f*)(in + i);
    v4f b = *(const v4f*)(in + i + 4);
    v8bf o;
#pragma unroll
    for (int j = 0; j < 4; ++j) { o[j] = f2bf(a[j]); o[4 + j] = f2bf(b[j]); }
    *(v8bf*)(out + i) = o;
}

// ---------------------------------------------------------------------------
// W[K][N] fp32 -> Wt[N][K] bf16 (32x32 tiles via LDS, coalesced both sides)
// ---------------------------------------------------------------------------
__global__ __launch_bounds__(256)
void transpose_cvt_kernel(const float* __restrict__ W, bf16_t* __restrict__ Wt,
                          int K, int N)
{
    __shared__ float tile[32][33];
    const int n0 = blockIdx.x * 32;
    const int k0 = blockIdx.y * 32;
    const int tx = threadIdx.x & 31;
    const int ty = threadIdx.x >> 5;     // 0..7
#pragma unroll
    for (int i = 0; i < 4; ++i)
        tile[ty + 8 * i][tx] = W[(long)(k0 + ty + 8 * i) * N + n0 + tx];
    __syncthreads();
#pragma unroll
    for (int i = 0; i < 4; ++i)
        Wt[(long)(n0 + ty + 8 * i) * K + k0 + tx] = f2bf(tile[tx][ty + 8 * i]);
}

// ---------------------------------------------------------------------------
// one K-step of WMMA work for a wave: 2x4 tiles of 16x16, K=32
// ---------------------------------------------------------------------------
__device__ __forceinline__ void wmmaStage(const bf16_t* __restrict__ cA,
                                          const bf16_t* __restrict__ cB,
                                          int waveM, int waveN,
                                          int halfK, int l16,
                                          v8f acc[2][4])
{
    // A 16x32 fragment: lanes 0-15 hold K 0-7 & 16-23; lanes 16-31: 8-15 & 24-31
    v16bf afrag[2];
#pragma unroll
    for (int mt = 0; mt < 2; ++mt) {
        int row = waveM + mt * 16 + l16;
        v8bf lo = *(const v8bf*)&cA[row * LDT + halfK * 8];
        v8bf hi = *(const v8bf*)&cA[row * LDT + 16 + halfK * 8];
        afrag[mt] = __builtin_shufflevector(lo, hi,
            0,1,2,3,4,5,6,7,8,9,10,11,12,13,14,15);
    }
    // B 32x16 fragment: lanes 0-15 hold K 0-7 & 16-23 of col; lanes 16-31: 8-15 & 24-31
    v16bf bfrag[4];
#pragma unroll
    for (int nt = 0; nt < 4; ++nt) {
        int col = waveN + nt * 16 + l16;
        v8bf lo = *(const v8bf*)&cB[col * LDT + halfK * 8];
        v8bf hi = *(const v8bf*)&cB[col * LDT + 16 + halfK * 8];
        bfrag[nt] = __builtin_shufflevector(lo, hi,
            0,1,2,3,4,5,6,7,8,9,10,11,12,13,14,15);
    }
#pragma unroll
    for (int mt = 0; mt < 2; ++mt)
#pragma unroll
        for (int nt = 0; nt < 4; ++nt)
            acc[mt][nt] = __builtin_amdgcn_wmma_f32_16x16x32_bf16(
                false, afrag[mt], false, bfrag[nt],
                (short)0, acc[mt][nt], false, false);
}

// ---------------------------------------------------------------------------
// GEMM: Out[M][N] = relu(A[M][K] @ Bt[N][K]^T + bias[N]), all bf16 operands,
// fp32 accumulate via v_wmma_f32_16x16x32_bf16.
// Block tile 128x128, 8 waves (4 M x 2 N), wave tile 32x64 (2x4 WMMA tiles).
// Branch-free software pipeline: double-buffered LDS fed by
// GLOBAL_LOAD_ASYNC_TO_LDS_B128, 2 K-steps unrolled per loop iteration so
// buffer parity is compile-time constant (no cndmask / per-iter branches).
// Requires M%128==0, N%128==0, K%64==0 (K/32 even).
// ---------------------------------------------------------------------------
__global__ __launch_bounds__(256)
void gemm_relu_async_kernel(const bf16_t* __restrict__ A,
                            const bf16_t* __restrict__ Bt,
                            const float* __restrict__ bias,
                            bf16_t* __restrict__ Out,
                            int M, int N, int K)
{
    __shared__ __align__(16) bf16_t sA[2][128 * LDT];
    __shared__ __align__(16) bf16_t sB[2][128 * LDT];

    const int tid   = threadIdx.x;
    const int lane  = tid & 31;
    const int wave  = tid >> 5;
    const int waveM = (wave & 3) * 32;
    const int waveN = (wave >> 2) * 64;
    const int mBlk  = blockIdx.y * 128;
    const int nBlk  = blockIdx.x * 128;

    // ---- per-thread async copy slots: tile = 128 rows x 4 B128-chunks
    const int v0 = tid;                  // chunk ids tid and tid+256
    const int v1 = tid + 256;
    const int rA0 = v0 >> 2, cA0 = (v0 & 3) * 8;
    const int rA1 = v1 >> 2, cA1 = (v1 & 3) * 8;

    const uint64_t gA0 = (uint64_t)(uintptr_t)(A  + (long)(mBlk + rA0) * K + cA0);
    const uint64_t gA1 = (uint64_t)(uintptr_t)(A  + (long)(mBlk + rA1) * K + cA1);
    const uint64_t gB0 = (uint64_t)(uintptr_t)(Bt + (long)(nBlk + rA0) * K + cA0);
    const uint64_t gB1 = (uint64_t)(uintptr_t)(Bt + (long)(nBlk + rA1) * K + cA1);

    unsigned lA0[2], lA1[2], lB0[2], lB1[2];
#pragma unroll
    for (int s = 0; s < 2; ++s) {
        lA0[s] = ldsOff(&sA[s][rA0 * LDT + cA0]);
        lA1[s] = ldsOff(&sA[s][rA1 * LDT + cA1]);
        lB0[s] = ldsOff(&sB[s][rA0 * LDT + cA0]);
        lB1[s] = ldsOff(&sB[s][rA1 * LDT + cA1]);
    }

#define ISSUE_STAGE(par, kb)                       \
    do {                                           \
        asyncCopyB128(lA0[par], gA0 + (kb));       \
        asyncCopyB128(lA1[par], gA1 + (kb));       \
        asyncCopyB128(lB0[par], gB0 + (kb));       \
        asyncCopyB128(lB1[par], gB1 + (kb));       \
    } while (0)

    v8f acc[2][4];
    const v8f vzero = {0.f,0.f,0.f,0.f,0.f,0.f,0.f,0.f};
#pragma unroll
    for (int mt = 0; mt < 2; ++mt)
#pragma unroll
        for (int nt = 0; nt < 4; ++nt) acc[mt][nt] = vzero;

    const int halfK  = lane >> 4;
    const int l16    = lane & 15;
    const int nSteps = K / 32;           // even

    // ---- prologue: stage 0 in flight into buffer 0
    ISSUE_STAGE(0, 0);
    uint64_t kb = 64;                    // stage byte offset (32 elems * 2B)

    // ---- steady state: two stages per iteration, fixed buffer parity
    for (int it = 0; it < nSteps - 2; it += 2) {
        ISSUE_STAGE(1, kb); kb += 64;
        asm volatile("s_wait_asynccnt 0x4" ::: "memory");
        __syncthreads();
        wmmaStage(sA[0], sB[0], waveM, waveN, halfK, l16, acc);
        __syncthreads();

        ISSUE_STAGE(0, kb); kb += 64;
        asm volatile("s_wait_asynccnt 0x4" ::: "memory");
        __syncthreads();
        wmmaStage(sA[1], sB[1], waveM, waveN, halfK, l16, acc);
        __syncthreads();
    }

    // ---- epilogue of pipeline: last two stages
    ISSUE_STAGE(1, kb);
    asm volatile("s_wait_asynccnt 0x4" ::: "memory");
    __syncthreads();
    wmmaStage(sA[0], sB[0], waveM, waveN, halfK, l16, acc);
    __syncthreads();
    asm volatile("s_wait_asynccnt 0x0" ::: "memory");
    __syncthreads();
    wmmaStage(sA[1], sB[1], waveM, waveN, halfK, l16, acc);

#undef ISSUE_STAGE

    // ---- epilogue: bias + relu, store bf16
    // C/D: VGPR r = row r (lanes 0-15) / row 8+r (lanes 16-31), col = lane%16
#pragma unroll
    for (int mt = 0; mt < 2; ++mt) {
#pragma unroll
        for (int nt = 0; nt < 4; ++nt) {
            int col = nBlk + waveN + nt * 16 + l16;
            float bv = bias[col];
#pragma unroll
            for (int r = 0; r < 8; ++r) {
                int row = mBlk + waveM + mt * 16 + halfK * 8 + r;
                float v = acc[mt][nt][r] + bv;
                v = v > 0.f ? v : 0.f;
                Out[(long)row * N + col] = f2bf(v);
            }
        }
    }
}

// ---------------------------------------------------------------------------
// Layer 3 + per-chunk mean: one block (128 thr) per chunk of 128 tokens.
// ---------------------------------------------------------------------------
__global__ __launch_bounds__(128)
void chunk_logits_kernel(const bf16_t* __restrict__ H2,
                         const float* __restrict__ W3,
                         const float* __restrict__ B3,
                         float* __restrict__ chunkLogits,
                         int K, int E)
{
    __shared__ float sW[512 * 8];
    __shared__ float sRed[8 * 128];
    const int tid   = threadIdx.x;
    const int chunk = blockIdx.x;

    for (int i = tid; i < K * E; i += 128) sW[i] = W3[i];
    __syncthreads();

    const bf16_t* h = H2 + ((long)chunk * CHUNK_SZ + tid) * K;
    float acc[8] = {0.f,0.f,0.f,0.f,0.f,0.f,0.f,0.f};
    for (int k = 0; k < K; k += 8) {
        v8bf hv = *(const v8bf*)(h + k);
#pragma unroll
        for (int i = 0; i < 8; ++i) {
            float x = (float)hv[i];
            const float* wr = &sW[(k + i) * 8];
#pragma unroll
            for (int e = 0; e < 8; ++e) acc[e] += x * wr[e];
        }
    }
#pragma unroll
    for (int e = 0; e < 8; ++e) sRed[e * 128 + tid] = acc[e];
    __syncthreads();
    for (int s = 64; s > 0; s >>= 1) {
        if (tid < s) {
#pragma unroll
            for (int e = 0; e < 8; ++e)
                sRed[e * 128 + tid] += sRed[e * 128 + tid + s];
        }
        __syncthreads();
    }
    if (tid < E)
        chunkLogits[(long)chunk * E + tid] =
            sRed[tid * 128] * (1.0f / CHUNK_SZ) + B3[tid];
}

// ---------------------------------------------------------------------------
// Hysteresis scan + stats + small outputs. Single block, 128 threads.
// ---------------------------------------------------------------------------
__global__ __launch_bounds__(128)
void finalize_kernel(const float* __restrict__ chunkLogits,  // [B*C][E]
                     int* __restrict__ expertIdx,            // ws, [B*C]
                     float* __restrict__ out,
                     int B, int C, int E,
                     long offIdx, long offCL, long offEnt,
                     long offUtil, long offFlip, long offConc)
{
    __shared__ float cl[128 * 8];
    __shared__ int   idx[128];
    __shared__ int   flipsTot;
    __shared__ float entSum;
    __shared__ float util[8];
    const int tid = threadIdx.x;
    const int BC  = B * C;

    for (int i = tid; i < BC * E; i += 128) cl[i] = chunkLogits[i];
    if (tid == 0) { flipsTot = 0; entSum = 0.f; }
    if (tid < 8) util[tid] = 0.f;
    __syncthreads();

    if (tid < B) {
        int prev = 0, flips = 0;
        for (int c = 0; c < C; ++c) {
            const float* l = &cl[(tid * C + c) * E];
            int top = 0; float best = l[0];
            for (int e = 1; e < E; ++e)
                if (l[e] > best) { best = l[e]; top = e; }
            int fin;
            if (c == 0) fin = top;
            else {
                bool sw = (l[top] - l[prev]) > TAU_C;
                if (sw) flips++;
                fin = sw ? top : prev;
            }
            prev = fin;
            idx[tid * C + c] = fin;
        }
        atomicAdd(&flipsTot, flips);
    }
    __syncthreads();

    if (tid < BC) {
        const float* l = &cl[tid * E];
        float mx = l[0];
        for (int e = 1; e < E; ++e) mx = fmaxf(mx, l[e]);
        float s = 0.f, ex[8];
        for (int e = 0; e < E; ++e) { ex[e] = __expf(l[e] - mx); s += ex[e]; }
        float ent = 0.f;
        for (int e = 0; e < E; ++e) {
            float p = ex[e] / s;
            ent -= p * logf(p + 1e-8f);
        }
        atomicAdd(&entSum, ent);
        atomicAdd(&util[idx[tid]], 1.0f / (float)BC);
    }
    __syncthreads();

    for (int i = tid; i < BC * E; i += 128) out[offCL + i] = cl[i];
    if (tid < BC) {
        out[offIdx + tid] = (float)idx[tid];
        expertIdx[tid]    = idx[tid];
    }
    if (tid == 0) {
        out[offEnt] = entSum / (float)BC;
        float nrm = 0.f;
        for (int e = 0; e < E; ++e) { out[offUtil + e] = util[e]; nrm += util[e] * util[e]; }
        out[offFlip] = (float)flipsTot / (float)(B * (C - 1));
        out[offConc] = sqrtf(nrm);
    }
}

// ---------------------------------------------------------------------------
// routing_weights[B,S,E] = one_hot broadcast. One thread per token.
// ---------------------------------------------------------------------------
__global__ __launch_bounds__(256)
void routing_weights_kernel(const int* __restrict__ expertIdx,
                            float* __restrict__ outRW,
                            int S, int C, int E)
{
    long t = (long)blockIdx.x * 256 + threadIdx.x;   // token over B*S
    int b = (int)(t / S);
    int s = (int)(t % S);
    int e = expertIdx[b * C + s / CHUNK_SZ];
    float* p = outRW + t * E;
#pragma unroll
    for (int j = 0; j < 8; ++j) p[j] = (j == e) ? 1.0f : 0.0f;
}

// ---------------------------------------------------------------------------
extern "C" void kernel_launch(void* const* d_in, const int* in_sizes, int n_in,
                              void* d_out, int out_size, void* d_ws, size_t ws_size,
                              hipStream_t stream)
{
    const float* x  = (const float*)d_in[0];
    // d_in[1] = prev_expert_indices: unused (scan init is zeros, first chunk forced)
    const float* w1 = (const float*)d_in[2];
    const float* b1 = (const float*)d_in[3];
    const float* w2 = (const float*)d_in[4];
    const float* b2 = (const float*)d_in[5];
    const float* w3 = (const float*)d_in[6];
    const float* b3 = (const float*)d_in[7];

    const int  B   = in_sizes[1];            // 4
    const int  H   = in_sizes[3];            // 1024
    const int  H2  = in_sizes[5];            // 512
    const int  D   = in_sizes[2] / H;        // 2048
    const long tok = (long)in_sizes[0] / D;  // B*S = 16384
    const int  S   = (int)(tok / B);         // 4096
    const int  E   = in_sizes[7];            // 8
    const int  C   = S / CHUNK_SZ;           // 32

    // workspace layout (bf16 elements unless noted)
    bf16_t* xb  = (bf16_t*)d_ws;             // [tok][D]
    bf16_t* w1t = xb  + tok * D;             // [H][D]   (transposed)
    bf16_t* w2t = w1t + (long)D * H;         // [H2][H]  (transposed)
    bf16_t* h1  = w2t + (long)H * H2;        // [tok][H]
    bf16_t* h2  = h1  + tok * H;             // [tok][H2]
    float*  cl  = (float*)(h2 + tok * H2);   // [B*C][E]
    int*    eix = (int*)(cl + (long)B * C * E);

    float* out = (float*)d_out;
    const long offRW   = 0;
    const long offIdx  = offRW  + (long)B * S * E;
    const long offCL   = offIdx + (long)B * C;
    const long offEnt  = offCL  + (long)B * C * E;
    const long offUtil = offEnt + 1;
    const long offFlip = offUtil + E;
    const long offConc = offFlip + 1;

    // ---- precision hoist: x -> bf16, w1/w2 -> transposed bf16
    cvt_bf16_kernel<<<(int)((tok * D) / 2048), 256, 0, stream>>>(x, xb, tok * D);
    transpose_cvt_kernel<<<dim3(H / 32,  D / 32), 256, 0, stream>>>(w1, w1t, D, H);
    transpose_cvt_kernel<<<dim3(H2 / 32, H / 32), 256, 0, stream>>>(w2, w2t, H, H2);

    // ---- Layer 1: h1 = relu(xb @ w1 + b1)
    gemm_relu_async_kernel<<<dim3(H / 128, (int)(tok / 128)), 256, 0, stream>>>(
        xb, w1t, b1, h1, (int)tok, H, D);
    // ---- Layer 2: h2 = relu(h1 @ w2 + b2)
    gemm_relu_async_kernel<<<dim3(H2 / 128, (int)(tok / 128)), 256, 0, stream>>>(
        h1, w2t, b2, h2, (int)tok, H2, H);
    // ---- Layer 3 + chunk mean
    chunk_logits_kernel<<<B * C, 128, 0, stream>>>(h2, w3, b3, cl, H2, E);
    // ---- Scan + stats + scalar outputs
    finalize_kernel<<<1, 128, 0, stream>>>(cl, eix, out, B, C, E,
                                           offIdx, offCL, offEnt,
                                           offUtil, offFlip, offConc);
    // ---- One-hot routing weights
    routing_weights_kernel<<<(int)(tok / 256), 256, 0, stream>>>(
        eix, out + offRW, S, C, E);
}
